// AnisotropicEdgeFilter_87454124081339
// MI455X (gfx1250) — matrix operand order repaired and checked by reading, not compiled, not run on hardware
//
#include <hip/hip_runtime.h>

// MI455X / gfx1250, wave32. Fused edge-filter MLP, single-pass:
//   h = elu(pos @ W1 + b1)            [E,128]   (per-lane VALU, builds WMMA A-frags in-register)
//   ew = h @ W2 + b2                  [E,8,32]  (v_wmma_f32_16x16x32_f16; bias folded into C-init)
//   out = einsum('ei,eio->eo', a, ew) [E,32]    (8 FMAs per WMMA n-tile, fused epilogue)
// Full f16-transposed W2 lives in LDS (69.6 KB; gfx1250 WGP has 320 KB LDS).
// A compiler memory barrier per tile stops LICM from hoisting LDS reads into
// registers (round 2 showed that causes scratch spills inside the WMMA chain).

typedef __attribute__((ext_vector_type(16))) _Float16 v16h;
typedef __attribute__((ext_vector_type(8)))  _Float16 v8h;
typedef __attribute__((ext_vector_type(8)))  float    v8f;
typedef __attribute__((ext_vector_type(4)))  float    v4f;
typedef __attribute__((ext_vector_type(2)))  float    v2f;

#define NEDGE   500000
#define NTILES  31250      // NEDGE / 16, exact
#define HIDDEN  128
#define NCOL    256        // IN_SIZE * OUT_SIZE
#define W2PAD   136        // halves per LDS row: 128 + 8 pad (272 B) -> conflict-free b128 reads
#define GRIDX   512

__global__ __launch_bounds__(128)
__attribute__((amdgpu_waves_per_eu(4)))
void aef_wmma_kernel(
    const float* __restrict__ edge_attr,   // [E,8]
    const float* __restrict__ edge_pos,    // [E,6]
    const float* __restrict__ W1,          // [6,128]
    const float* __restrict__ b1,          // [128]
    const float* __restrict__ W2,          // [128,256]
    const float* __restrict__ b2,          // [256]
    float* __restrict__ out)               // [E,32]
{
    // ---- LDS: 69632 + 3072 + 512 + 1024 + 2048 = 76288 B (< 320 KB WGP LDS) ----
    __shared__ __align__(16) _Float16 s_w2[NCOL * W2PAD]; // W2 transposed [N][K], f16
    __shared__ __align__(16) float    s_w1[6 * 128];
    __shared__ __align__(16) float    s_b1[128];
    __shared__ __align__(16) float    s_b2[NCOL];
    __shared__ __align__(16) float    s_attr[4][16 * 8];  // per-wave edge_attr tile

    const int tid = threadIdx.x;

    // ---- stage W1 / b1 ----
    #pragma unroll
    for (int idx = tid; idx < 6 * 128; idx += 128) s_w1[idx] = W1[idx];
    s_b1[tid] = b1[tid];

    // ---- stage + transpose + f16-convert full W2 (row N of s_w2 = column N of W2) ----
    #pragma unroll
    for (int rr = 0; rr < 2; ++rr) {
        const int N = tid + rr * 128;              // 0..255
        s_b2[N] = b2[N];
        _Float16* dst = &s_w2[N * W2PAD];
        #pragma unroll
        for (int kb = 0; kb < 16; ++kb) {          // 128 K in chunks of 8 -> ds_store_b128
            v8h h8;
            #pragma unroll
            for (int j = 0; j < 8; ++j)
                h8[j] = (_Float16)W2[(kb * 8 + j) * NCOL + N];
            *(v8h*)(dst + kb * 8) = h8;
        }
    }
    __syncthreads();

    const int wave = tid >> 5;
    const int lane = tid & 31;
    const int hf   = lane >> 4;          // lane half (A/B fragment K-run selector)
    const int lrow = lane & 15;          // M row (A/D) == N col (B/D) within 16-tile
    float* attr_w  = s_attr[wave];

    for (int tile = blockIdx.x * 4 + wave; tile < NTILES; tile += gridDim.x * 4) {
        const int e0 = tile * 16;

        // Block cross-iteration LICM of LDS reads (W1 / W2 / b1 / b2): re-reading
        // LDS per tile is broadcast-cheap; hoisting them spills to scratch.
        asm volatile("" ::: "memory");

        // prefetch next tile's edge data (global_prefetch_b8)
        {
            const int nt = tile + gridDim.x * 4;
            if (nt < NTILES) {
                __builtin_prefetch(edge_attr + (size_t)nt * 16 * 8, 0, 1);
                __builtin_prefetch(edge_pos  + (size_t)nt * 16 * 6, 0, 1);
            }
        }

        // ---- stage edge_attr tile (16x8 f32 = 512 B): one b128 per lane ----
        {
            v4f a4 = *(const v4f*)(edge_attr + (size_t)e0 * 8 + lane * 4);
            *(v4f*)(attr_w + lane * 4) = a4;
        }

        // ---- per-lane pos row (m = lrow); 24 B = 3x b64 loads ----
        float p[6];
        {
            const float* pp = edge_pos + (size_t)(e0 + lrow) * 6;
            v2f p01 = *(const v2f*)(pp);
            v2f p23 = *(const v2f*)(pp + 2);
            v2f p45 = *(const v2f*)(pp + 4);
            p[0] = p01.x; p[1] = p01.y; p[2] = p23.x;
            p[3] = p23.y; p[4] = p45.x; p[5] = p45.y;
        }

        // ---- GEMM1 + ELU, producing A-fragments directly in WMMA layout ----
        // A 16x32 f16 layout: lanes 0-15 hold M=lane, K = 32s+{0..7,16..23};
        // lanes 16-31 hold M=lane-16, K = 32s+{8..15,24..31}. cb = hf*8 selects the run.
        v16h afrag[4];
        const int cb = hf * 8;
        #pragma unroll
        for (int s = 0; s < 4; ++s) {
            #pragma unroll
            for (int run = 0; run < 2; ++run) {
                const int c0 = 32 * s + 16 * run + cb;    // half-wave uniform -> LDS broadcast
                v4f h0 = *(const v4f*)(&s_b1[c0]);
                v4f h1 = *(const v4f*)(&s_b1[c0 + 4]);
                #pragma unroll
                for (int k = 0; k < 6; ++k) {
                    v4f w0 = *(const v4f*)(&s_w1[k * 128 + c0]);
                    v4f w1 = *(const v4f*)(&s_w1[k * 128 + c0 + 4]);
                    h0 += p[k] * w0;
                    h1 += p[k] * w1;
                }
                #pragma unroll
                for (int j = 0; j < 4; ++j) {
                    float x0 = h0[j];
                    float y0 = x0 > 0.f ? x0 : (__expf(x0) - 1.f);   // ELU
                    afrag[s][run * 8 + j] = (_Float16)y0;
                    float x1 = h1[j];
                    float y1 = x1 > 0.f ? x1 : (__expf(x1) - 1.f);
                    afrag[s][run * 8 + 4 + j] = (_Float16)y1;
                }
            }
        }

        // ---- GEMM2: 16 n-tiles = (i: input channel 0..7) x (pq: o-half 0..1) ----
        float acc[2][8];
        #pragma unroll
        for (int r = 0; r < 8; ++r) { acc[0][r] = 0.f; acc[1][r] = 0.f; }

        #pragma unroll
        for (int i = 0; i < 8; ++i) {
            // einsum scalars a[e0 + r + 8*hf][i]: shared by both o-halves, load once
            float av[8];
            #pragma unroll
            for (int r = 0; r < 8; ++r)
                av[r] = attr_w[(r + 8 * hf) * 8 + i];

            #pragma unroll
            for (int pq = 0; pq < 2; ++pq) {           // o-half parity (compile-time)
                const int n = 2 * i + pq;              // n-tile: N in [16n, 16n+16)
                // C init = b2 broadcast: C[m][t] = b2[16n + t], same for all 8 row-regs
                float b2v = s_b2[n * 16 + lrow];
                v8f d;
                #pragma unroll
                for (int r = 0; r < 8; ++r) d[r] = b2v;

                #pragma unroll
                for (int s = 0; s < 4; ++s) {
                    // B 32x16 f16: lane holds K = 32s+8hf+{0..7} and 32s+16+8hf+{0..7} for column N = 16n+lrow;
                    // stored K-contiguous -> two ds_load_b128
                    const _Float16* bp = &s_w2[(n * 16 + lrow) * W2PAD + 32 * s + 8 * hf];
                    v8h blo = *(const v8h*)bp;
                    v8h bhi = *(const v8h*)(bp + 16);
                    v16h bf = __builtin_shufflevector(blo, bhi,
                        0, 1, 2, 3, 4, 5, 6, 7, 8, 9, 10, 11, 12, 13, 14, 15);
                    d = __builtin_amdgcn_wmma_f32_16x16x32_f16(
                            false, afrag[s], false, bf, (short)0, d, false, false);
                }
                #pragma unroll
                for (int r = 0; r < 8; ++r)
                    acc[pq][r] += av[r] * d[r];
            }
        }

        // ---- store out[e0+m][o], m = r + 8*hf; o = lrow (pq=0) and 16+lrow (pq=1) ----
        float* op = out + (size_t)e0 * 32 + lrow;
        #pragma unroll
        for (int r = 0; r < 8; ++r) {
            op[(size_t)(r + 8 * hf) * 32]      = acc[0][r];
            op[(size_t)(r + 8 * hf) * 32 + 16] = acc[1][r];
        }
    }
}

extern "C" void kernel_launch(void* const* d_in, const int* in_sizes, int n_in,
                              void* d_out, int out_size, void* d_ws, size_t ws_size,
                              hipStream_t stream) {
    const float* edge_attr = (const float*)d_in[0];
    const float* edge_pos  = (const float*)d_in[1];
    const float* W1        = (const float*)d_in[2];
    const float* b1        = (const float*)d_in[3];
    const float* W2        = (const float*)d_in[4];
    const float* b2        = (const float*)d_in[5];
    float*       out       = (float*)d_out;
    (void)in_sizes; (void)n_in; (void)out_size; (void)d_ws; (void)ws_size;

    dim3 grid(GRIDX, 1);
    dim3 block(128);       // 4 wave32 waves; each wave owns a 16-edge WMMA tile
    aef_wmma_kernel<<<grid, block, 0, stream>>>(edge_attr, edge_pos, W1, b1, W2, b2, out);
}
